// ICCA_67912022884667
// MI455X (gfx1250) — matrix operand order repaired and checked by reading, not compiled, not run on hardware
//
#include <hip/hip_runtime.h>
#include <hip/hip_bf16.h>
#include <stdint.h>

#define NB   16
#define LBL  24
#define DIM  1024
#define HIDN 64
#define ALPHA 0.5f
#define EPSV 1e-8f
#define FCIN (128*56*56)          // 401408
#define SPLITS 256
#define KSPLIT (FCIN/SPLITS)      // 1568
static_assert(KSPLIT*SPLITS == FCIN, "split");

typedef __bf16 bf16;
typedef __attribute__((ext_vector_type(16))) __bf16 bfx16;
typedef __attribute__((ext_vector_type(8)))  float  fx8;
typedef __attribute__((ext_vector_type(2)))  float  fx2;
typedef int vi4 __attribute__((__vector_size__(4 * sizeof(int))));

// -------- CDNA5 async global->LDS (guarded; falls back to plain LDS stores) --------
#if defined(__AMDGCN__) && __has_builtin(__builtin_amdgcn_global_load_async_to_lds_b128) && __has_builtin(__builtin_amdgcn_s_wait_asynccnt)
#define USE_ASYNC_LDS 1
#else
#define USE_ASYNC_LDS 0
#endif

__device__ __forceinline__ void g2l_b128(void* l, const void* g) {
#if USE_ASYNC_LDS
  __builtin_amdgcn_global_load_async_to_lds_b128(
      (__attribute__((address_space(1))) vi4*)g,
      (__attribute__((address_space(3))) vi4*)l, 0, 0);
#else
  *(uint4*)l = *(const uint4*)g;
#endif
}
__device__ __forceinline__ void wait_async0() {
#if USE_ASYNC_LDS
  __builtin_amdgcn_s_wait_asynccnt(0);
#endif
}

// ---------------- conv1 + relu + maxpool2 -> bf16 [16,64,112,112] ----------------
__global__ void k_conv1(const float* __restrict__ img, const float* __restrict__ w,
                        const float* __restrict__ b, bf16* __restrict__ h1) {
  int idx = blockIdx.x * blockDim.x + threadIdx.x;
  const int TOT = NB * 64 * 112 * 112;
  if (idx >= TOT) return;
  int x = idx % 112; int t = idx / 112;
  int y = t % 112;   t /= 112;
  int c = t % 64;    int n = t / 64;
  float wv[27];
#pragma unroll
  for (int i = 0; i < 27; i++) wv[i] = w[c * 27 + i];
  float bias = b[c];
  float best = -INFINITY;
#pragma unroll
  for (int dy = 0; dy < 2; dy++)
#pragma unroll
  for (int dx = 0; dx < 2; dx++) {
    int Y = 2 * y + dy, X = 2 * x + dx;
    float acc = bias;
#pragma unroll
    for (int ic = 0; ic < 3; ic++)
#pragma unroll
    for (int ky = 0; ky < 3; ky++) {
      int iy = Y + ky - 1;
      if ((unsigned)iy >= 224u) continue;
      const float* row = img + (((size_t)n * 3 + ic) * 224 + iy) * 224;
#pragma unroll
      for (int kx = 0; kx < 3; kx++) {
        int ix = X + kx - 1;
        if ((unsigned)ix < 224u) acc = fmaf(row[ix], wv[ic * 9 + ky * 3 + kx], acc);
      }
    }
    acc = fmaxf(acc, 0.f);
    best = fmaxf(best, acc);
  }
  h1[idx] = (bf16)best;
}

// ---------------- conv2 weights fp32 -> bf16, layout [oc][ic*9] kept ----------------
__global__ void k_w2prep(const float* __restrict__ w, bf16* __restrict__ o) {
  int i = blockIdx.x * blockDim.x + threadIdx.x;
  if (i < 128 * 576) o[i] = (bf16)w[i];
}

// ---------------- conv2 implicit GEMM via v_wmma_f32_16x16x32_bf16 ----------------
// M = 16*112*112 rows (one 16-row tile per wave), N = 128 (8 tiles), K = 576.
// 112%16==0 so every M tile has fixed (image, oy) and a contiguous 16-wide ox range:
// the whole receptive field is a 64x3x18 bf16 halo tile staged once in LDS.
__global__ void __launch_bounds__(32) k_conv2(const bf16* __restrict__ h1,
                                              const bf16* __restrict__ w2,
                                              const float* __restrict__ bias,
                                              bf16* __restrict__ out) {
  __shared__ bf16 aT[64 * 3 * 18];   // [ic][dy][dx] halo tile, 6912 B
  __shared__ uint4 ldsB[512];        // B panel: 128 n x 32 k bf16, [n][k]
  const int L = threadIdx.x;
  const int mbase = blockIdx.x * 16;
  const int nimg = mbase / 12544;
  const int rem  = mbase - nimg * 12544;
  const int oy   = rem / 112;
  const int ox0  = rem - oy * 112;        // multiple of 16
  const int lx = L & 15;
  const int hiHalf = L >> 4;

  // stage zero-padded activation halo tile (plain LDS stores)
#pragma unroll 4
  for (int it = 0; it < 108; it++) {      // 3456 elements / 32 lanes
    int e = it * 32 + L;
    int ic = e / 54; int r = e - ic * 54;
    int dy = r / 18; int dx = r - dy * 18;
    int iy = oy + dy - 1, ix = ox0 + dx - 1;
    bf16 v = (bf16)0.f;
    if ((unsigned)iy < 112u && (unsigned)ix < 112u)
      v = h1[(((size_t)nimg * 64 + ic) * 112 + iy) * 112 + ix];
    aT[e] = v;
  }
  __syncthreads();

  fx8 acc[8];
#pragma unroll
  for (int t = 0; t < 8; t++)
#pragma unroll
    for (int i = 0; i < 8; i++) acc[t][i] = 0.f;

  for (int k0 = 0; k0 < 576; k0 += 32) {
    // 1) issue async B-panel load: w2[n][k0..k0+31] -> LDS (ASYNCcnt)
#pragma unroll
    for (int it = 0; it < 16; it++) {
      int chunk = it * 32 + L;            // 0..511
      int n = chunk >> 2; int c = chunk & 3;
      g2l_b128(&ldsB[chunk], w2 + (size_t)n * 576 + k0 + 8 * c);
    }

    // 2) build A fragment from LDS halo tile (overlaps the async loads)
    //    lane holds row M=lx; K = k0 + hiHalf*8 + {0..7} and +16
    union { bfx16 v; bf16 e[16]; } A;
    int kb = k0 + (hiHalf << 3);
#pragma unroll
    for (int j = 0; j < 8; j++) {
      int kk = kb + j;
      int ic = kk / 9; int r = kk - ic * 9; int ky = r / 3; int kx = r - ky * 3;
      A.e[j] = aT[ic * 54 + ky * 18 + lx + kx];
      kk += 16;
      ic = kk / 9; r = kk - ic * 9; ky = r / 3; kx = r - ky * 3;
      A.e[8 + j] = aT[ic * 54 + ky * 18 + lx + kx];
    }

    // 3) panel ready
    wait_async0();

    // 4) 8 N-tiles: B fragment = 2x ds_load_b128, then WMMA
#pragma unroll
    for (int t = 0; t < 8; t++) {
      int n = t * 16 + lx;
      int i0 = n * 4 + hiHalf;            // 16B chunk index: n*64B + half*16B
      union { uint4 q[2]; bfx16 v; } B;
      B.q[0] = ldsB[i0];
      B.q[1] = ldsB[i0 + 2];
      acc[t] = __builtin_amdgcn_wmma_f32_16x16x32_bf16(false, A.v, false, B.v,
                                                       (short)0, acc[t], false, false);
    }
  }

  // epilogue: bias + relu, write pre-pool bf16 [16,128,112,112]
#pragma unroll
  for (int t = 0; t < 8; t++) {
    int oc = t * 16 + lx;
    float bv = bias[oc];
#pragma unroll
    for (int r = 0; r < 8; r++) {
      int ox = ox0 + r + (hiHalf << 3);   // C layout: VGPR r -> M=r (+8 for upper half)
      float v = fmaxf(acc[t][r] + bv, 0.f);
      out[(((size_t)nimg * 128 + oc) * 112 + oy) * 112 + ox] = (bf16)v;
    }
  }
}

// ---------------- maxpool2 conv2 output -> pooled fp32 [16, 401408] ----------------
__global__ void k_pool2(const bf16* __restrict__ c2, float* __restrict__ pooled) {
  int idx = blockIdx.x * blockDim.x + threadIdx.x;
  const int TOT = NB * 128 * 56 * 56;
  if (idx >= TOT) return;
  int x = idx % 56; int t = idx / 56;
  int y = t % 56;   t /= 56;
  int c = t % 128;  int n = t / 128;
  size_t base = (((size_t)n * 128 + c) * 112 + 2 * y) * 112 + 2 * x;
  float m = (float)c2[base];
  m = fmaxf(m, (float)c2[base + 1]);
  m = fmaxf(m, (float)c2[base + 112]);
  m = fmaxf(m, (float)c2[base + 113]);
  pooled[(size_t)n * FCIN + c * 3136 + y * 56 + x] = m;
}

// ---------------- fc split-K GEMM via v_wmma_f32_16x16x4_f32 ----------------
// HBM-bound on 1.6 GB fp32 fc_w -> stream fp32 directly through the f32 tensor op.
__global__ void __launch_bounds__(32) k_fc(const float* __restrict__ pooled,
                                           const float* __restrict__ fcw,
                                           float* __restrict__ part) {
  const int L = threadIdx.x;
  const int s = blockIdx.x;
  const int g = blockIdx.y;
  const int kbeg = s * KSPLIT, kend = kbeg + KSPLIT;
  const int koff = (L >> 4) * 2;
  const int m = L & 15;
  const float* __restrict__ arow = pooled + (size_t)m * FCIN;

  fx8 acc[8];
  int nidx[8];
#pragma unroll
  for (int t = 0; t < 8; t++) {
    nidx[t] = (g * 8 + t) * 16 + m;
#pragma unroll
    for (int i = 0; i < 8; i++) acc[t][i] = 0.f;
  }

  for (int k0 = kbeg; k0 < kend; k0 += 4) {
    fx2 a; a.x = arow[k0 + koff]; a.y = arow[k0 + koff + 1];
    bool pf = ((k0 & 31) == 0);
#pragma unroll
    for (int t = 0; t < 8; t++) {
      const float* brow = fcw + (size_t)nidx[t] * FCIN;
      if (pf) __builtin_prefetch(brow + k0 + 128, 0, 0);
      fx2 b; b.x = brow[k0 + koff]; b.y = brow[k0 + koff + 1];
      acc[t] = __builtin_amdgcn_wmma_f32_16x16x4_f32(false, a, false, b,
                                                     (short)0, acc[t], false, false);
    }
  }
#pragma unroll
  for (int t = 0; t < 8; t++)
#pragma unroll
    for (int r = 0; r < 8; r++) {
      int mrow = r + ((L >> 4) << 3);
      part[((size_t)s * 16 + mrow) * 1024 + nidx[t]] = acc[t][r];
    }
}

__global__ void k_fcred(const float* __restrict__ part, const float* __restrict__ fcb,
                        float* __restrict__ imgf) {
  int i = blockIdx.x * blockDim.x + threadIdx.x;
  if (i >= NB * 1024) return;
  float a = fcb[i & 1023];
  for (int s = 0; s < SPLITS; s++) a += part[(size_t)s * NB * 1024 + i];
  imgf[i] = a;
}

// ---------------- text MLP ----------------
__global__ void k_mlp1(const float* __restrict__ txt, const float* __restrict__ w1,
                       const float* __restrict__ b1, float* __restrict__ hid) {
  int i = blockIdx.x * blockDim.x + threadIdx.x;
  if (i >= NB * 512) return;
  int j = i % 512, n = i / 512;
  float a = b1[j];
  for (int d = 0; d < DIM; d++) a = fmaf(txt[n * DIM + d], w1[(size_t)j * DIM + d], a);
  hid[i] = fmaxf(a, 0.f);
}
__global__ void k_mlp2(const float* __restrict__ hid, const float* __restrict__ w2,
                       const float* __restrict__ b2, float* __restrict__ txtf) {
  int i = blockIdx.x * blockDim.x + threadIdx.x;
  if (i >= NB * 1024) return;
  int o = i % 1024, n = i / 1024;
  float a = b2[o];
  for (int j = 0; j < 512; j++) a = fmaf(hid[n * 512 + j], w2[(size_t)o * 512 + j], a);
  txtf[i] = a;
}

// ---------------- label path ----------------
__global__ void k_lemb(const float* __restrict__ emb, const float* __restrict__ lw,
                       float* __restrict__ lemb) {
  int i = blockIdx.x * blockDim.x + threadIdx.x;
  if (i >= LBL * LBL) return;
  int j = i % LBL, r = i / LBL;
  float a = 0.f;
  for (int d = 0; d < DIM; d++) a = fmaf(emb[r * DIM + d], lw[j * DIM + d], a);
  lemb[i] = a;
}
__global__ void k_mlf(const int* __restrict__ ml, const float* __restrict__ lemb,
                      float* __restrict__ mlf) {
  int i = blockIdx.x * blockDim.x + threadIdx.x;
  if (i >= NB * LBL) return;
  int j = i % LBL, n = i / LBL;
  float a = 0.f;
  for (int l = 0; l < LBL; l++) a = fmaf((float)ml[n * LBL + l], lemb[l * LBL + j], a);
  mlf[i] = a;
}

// ---------------- adjacency: cos sims + scaled label co-occurrence ----------------
__global__ void k_adj(const float* __restrict__ imgf, const float* __restrict__ txtf,
                      const float* __restrict__ mlf, float* __restrict__ adj) {
  __shared__ float ni[16], nt[16];
  int t = threadIdx.x;            // 256 threads
  if (t < 16) {
    float s = 0.f;
    for (int d = 0; d < DIM; d++) { float v = imgf[t * DIM + d]; s = fmaf(v, v, s); }
    ni[t] = fmaxf(sqrtf(s), EPSV);
  } else if (t < 32) {
    int i = t - 16; float s = 0.f;
    for (int d = 0; d < DIM; d++) { float v = txtf[i * DIM + d]; s = fmaf(v, v, s); }
    nt[i] = fmaxf(sqrtf(s), EPSV);
  }
  __syncthreads();
  int i = t >> 4, j = t & 15;
  float di = 0.f, dt = 0.f, dm = 0.f;
  for (int d = 0; d < DIM; d++) {
    di = fmaf(imgf[i * DIM + d], imgf[j * DIM + d], di);
    dt = fmaf(txtf[i * DIM + d], txtf[j * DIM + d], dt);
  }
  for (int l = 0; l < LBL; l++) dm = fmaf(mlf[i * LBL + l], mlf[j * LBL + l], dm);
  adj[t] = di / (ni[i] * ni[j]) + dt / (nt[i] * nt[j]) + dm * (ALPHA / (float)LBL);
}

// ---------------- concat + GCN ----------------
#define XCAT (2 * DIM + LBL)      // 2072
__global__ void k_concat(const float* __restrict__ imgf, const float* __restrict__ txtf,
                         const float* __restrict__ mlf, float* __restrict__ x) {
  int i = blockIdx.x * blockDim.x + threadIdx.x;
  if (i >= NB * XCAT) return;
  int c = i % XCAT, n = i / XCAT;
  float v;
  if (c < DIM)            v = imgf[n * DIM + c];
  else if (c < 2 * DIM)   v = txtf[n * DIM + (c - DIM)];
  else                    v = mlf[n * LBL + (c - 2 * DIM)];
  x[i] = v;
}
__global__ void k_xw(const float* __restrict__ x, const float* __restrict__ w,
                     float* __restrict__ o, int K, int H) {
  int i = blockIdx.x * blockDim.x + threadIdx.x;
  if (i >= NB * H) return;
  int h = i % H, n = i / H;
  float a = 0.f;
  for (int k = 0; k < K; k++) a = fmaf(x[n * K + k], w[(size_t)k * H + h], a);
  o[i] = a;
}
__global__ void k_adjapply(const float* __restrict__ adj, const float* __restrict__ xin,
                           const float* __restrict__ b, float* __restrict__ o, int H) {
  int i = blockIdx.x * blockDim.x + threadIdx.x;
  if (i >= NB * H) return;
  int h = i % H, n = i / H;
  float a = b[h];
  for (int m = 0; m < NB; m++) a = fmaf(adj[n * NB + m], xin[m * H + h], a);
  o[i] = a;
}

extern "C" void kernel_launch(void* const* d_in, const int* in_sizes, int n_in,
                              void* d_out, int out_size, void* d_ws, size_t ws_size,
                              hipStream_t stream) {
  const float* image   = (const float*)d_in[0];
  const float* text    = (const float*)d_in[1];
  const int*   mlmat   = (const int*)  d_in[2];
  const float* conv1_w = (const float*)d_in[3];
  const float* conv1_b = (const float*)d_in[4];
  const float* conv2_w = (const float*)d_in[5];
  const float* conv2_b = (const float*)d_in[6];
  const float* fc_w    = (const float*)d_in[7];
  const float* fc_b    = (const float*)d_in[8];
  const float* mlp_w1  = (const float*)d_in[9];
  const float* mlp_b1  = (const float*)d_in[10];
  const float* mlp_w2  = (const float*)d_in[11];
  const float* mlp_b2  = (const float*)d_in[12];
  const float* lab_emb = (const float*)d_in[13];
  const float* lab_w   = (const float*)d_in[14];
  const float* gcn0_w  = (const float*)d_in[15];
  const float* gcn0_b  = (const float*)d_in[16];
  const float* gcn1_w  = (const float*)d_in[17];
  const float* gcn1_b  = (const float*)d_in[18];
  float* outp = (float*)d_out;

  char* ws = (char*)d_ws;
  size_t off = 0;
  auto alloc = [&](size_t bytes) { char* p = ws + off; off += (bytes + 255) & ~(size_t)255; return p; };
  bf16*  h1     = (bf16*) alloc((size_t)NB * 64 * 112 * 112 * 2);
  bf16*  w2b    = (bf16*) alloc(128 * 576 * 2);
  bf16*  c2     = (bf16*) alloc((size_t)NB * 128 * 112 * 112 * 2);
  float* pooled = (float*)alloc((size_t)NB * FCIN * 4);
  float* part   = (float*)alloc((size_t)SPLITS * NB * 1024 * 4);
  float* imgf   = (float*)alloc(NB * 1024 * 4);
  float* hid    = (float*)alloc(NB * 512 * 4);
  float* txtf   = (float*)alloc(NB * 1024 * 4);
  float* lemb   = (float*)alloc(LBL * LBL * 4);
  float* mlf    = (float*)alloc(NB * LBL * 4);
  float* adjm   = (float*)alloc(NB * NB * 4);
  float* xcat   = (float*)alloc(NB * XCAT * 4);
  float* xw0    = (float*)alloc(NB * HIDN * 4);
  float* g0     = (float*)alloc(NB * HIDN * 4);
  float* xw1    = (float*)alloc(NB * HIDN * 4);
  (void)ws_size; (void)in_sizes; (void)n_in; (void)out_size;

  const int B = 256;
  // image branch
  k_conv1<<<(NB * 64 * 112 * 112 + B - 1) / B, B, 0, stream>>>(image, conv1_w, conv1_b, h1);
  k_w2prep<<<(128 * 576 + B - 1) / B, B, 0, stream>>>(conv2_w, w2b);
  k_conv2<<<NB * 112 * 112 / 16, 32, 0, stream>>>(h1, w2b, conv2_b, c2);
  k_pool2<<<(NB * 128 * 56 * 56 + B - 1) / B, B, 0, stream>>>(c2, pooled);
  k_fc<<<dim3(SPLITS, 8), 32, 0, stream>>>(pooled, fc_w, part);
  k_fcred<<<(NB * 1024 + B - 1) / B, B, 0, stream>>>(part, fc_b, imgf);
  // text branch
  k_mlp1<<<(NB * 512 + B - 1) / B, B, 0, stream>>>(text, mlp_w1, mlp_b1, hid);
  k_mlp2<<<(NB * 1024 + B - 1) / B, B, 0, stream>>>(hid, mlp_w2, mlp_b2, txtf);
  // labels
  k_lemb<<<(LBL * LBL + B - 1) / B, B, 0, stream>>>(lab_emb, lab_w, lemb);
  k_mlf<<<(NB * LBL + B - 1) / B, B, 0, stream>>>(mlmat, lemb, mlf);
  // adjacency + GCN
  k_adj<<<1, 256, 0, stream>>>(imgf, txtf, mlf, adjm);
  k_concat<<<(NB * XCAT + B - 1) / B, B, 0, stream>>>(imgf, txtf, mlf, xcat);
  k_xw<<<(NB * HIDN + B - 1) / B, B, 0, stream>>>(xcat, gcn0_w, xw0, XCAT, HIDN);
  k_adjapply<<<(NB * HIDN + B - 1) / B, B, 0, stream>>>(adjm, xw0, gcn0_b, g0, HIDN);
  k_xw<<<(NB * HIDN + B - 1) / B, B, 0, stream>>>(g0, gcn1_w, xw1, HIDN, HIDN);
  k_adjapply<<<(NB * HIDN + B - 1) / B, B, 0, stream>>>(adjm, xw1, gcn1_b, outp, HIDN);
}
